// Encoder_88364657148157
// MI455X (gfx1250) — matrix operand order repaired and checked by reading, not compiled, not run on hardware
//
#include <hip/hip_runtime.h>
#include <hip/hip_bf16.h>

typedef __bf16 bf16_t;
typedef __attribute__((ext_vector_type(16))) __bf16 v16bf;
typedef __attribute__((ext_vector_type(8)))  __bf16 v8bf;
typedef __attribute__((ext_vector_type(8)))  float  v8f;
typedef __attribute__((ext_vector_type(4)))  int    v4i;

#define N_NODES 50000
#define N_EDGES 800000

#if defined(__HIP_DEVICE_COMPILE__) && __has_builtin(__builtin_amdgcn_global_load_async_to_lds_b128) && __has_builtin(__builtin_amdgcn_s_wait_asynccnt)
#define USE_ASYNC 1
#else
#define USE_ASYNC 0
#endif

__device__ __forceinline__ v8bf ld8(const bf16_t* p){ return *(const v8bf*)p; }

__device__ __forceinline__ v16bf mkfrag(v8bf lo, v8bf hi){
  v16bf r;
#pragma unroll
  for (int i = 0; i < 8; ++i){ r[i] = lo[i]; r[i+8] = hi[i]; }
  return r;
}

__device__ __forceinline__ v8f wmma_bf16(v16bf a, v16bf b, v8f c){
  return __builtin_amdgcn_wmma_f32_16x16x32_bf16(false, a, false, b, (short)0, c, false, false);
}

__device__ __forceinline__ float siluf(float x){ return x / (1.0f + __expf(-x)); }
__device__ __forceinline__ float geluf(float x){
  float u = 0.7978845608028654f * (x + 0.044715f * x * x * x);
  return 0.5f * x * (1.0f + tanhf(u));
}

#if USE_ASYNC
typedef __attribute__((address_space(1))) v4i* gv4i_p;
typedef __attribute__((address_space(3))) v4i* lv4i_p;
// 16-byte async copy global -> LDS (one lane's share of a global_load_async_to_lds_b128)
__device__ __forceinline__ void async_cp16(const bf16_t* g, bf16_t* l){
  gv4i_p gp = (gv4i_p)(unsigned long long)(const void*)g;
  lv4i_p lp = (lv4i_p)(unsigned int)(unsigned long long)(void*)l;
  __builtin_amdgcn_global_load_async_to_lds_b128(gp, lp, 0, 0);
}
#endif

// ---------------- prep kernels ----------------

// W [K, Nout] f32 row-major  ->  Bt [Nout, Kpad] bf16 (zero-padded K)
__global__ void k_wcvt(const float* __restrict__ w, bf16_t* __restrict__ bt,
                       int K, int Kpad, int Nout){
  int idx = blockIdx.x * blockDim.x + threadIdx.x;
  if (idx >= Kpad * Nout) return;
  int n = idx / Kpad, kp = idx % Kpad;
  bt[idx] = (kp < K) ? (bf16_t)w[kp * Nout + n] : (bf16_t)0.0f;
}

__global__ void k_pad_nodes(const float* __restrict__ na, bf16_t* __restrict__ xp){
  int idx = blockIdx.x * blockDim.x + threadIdx.x;
  if (idx >= N_NODES * 96) return;
  int n = idx / 96, k = idx % 96;
  xp[idx] = (k < 74) ? (bf16_t)na[n * 74 + k] : (bf16_t)0.0f;
}

__global__ void k_tail(const float* __restrict__ xyz, const float* __restrict__ ea,
                       const int* __restrict__ src, const int* __restrict__ dst,
                       bf16_t* __restrict__ tail){
  int e = blockIdx.x * blockDim.x + threadIdx.x;
  if (e >= N_EDGES) return;
  int s = src[e], d = dst[e];
  float dx = xyz[s*3+0] - xyz[d*3+0];
  float dy = xyz[s*3+1] - xyz[d*3+1];
  float dz = xyz[s*3+2] - xyz[d*3+2];
  tail[e*8+0] = (bf16_t)(dx*dx + dy*dy + dz*dz);
#pragma unroll
  for (int i = 0; i < 7; ++i) tail[e*8+1+i] = (bf16_t)ea[e*7+i];
}

__global__ void k_zero_f32(float* __restrict__ p, long n){
  long i = (long)blockIdx.x * blockDim.x + threadIdx.x;
  if (i < n) p[i] = 0.0f;
}

__global__ void k_f32_to_bf16(const float* __restrict__ in, bf16_t* __restrict__ out, long n){
  long i = (long)blockIdx.x * blockDim.x + threadIdx.x;
  if (i < n) out[i] = (bf16_t)in[i];
}

// ---------------- 1-wave GEMM kernels (16 rows x 128 cols per wave) ----------------

__global__ __launch_bounds__(32)
void k_gemm_ln_gelu(const bf16_t* __restrict__ A, int Kpad,
                    const bf16_t* __restrict__ Bt, const float* __restrict__ bias,
                    const float* __restrict__ g, const float* __restrict__ bb,
                    bf16_t* __restrict__ out){
  __shared__ float smf[16*128];
  int l = threadIdx.x, r = l & 15, half = l >> 4;
  long t = blockIdx.x;
  const bf16_t* arow = A + (t*16 + r) * (long)Kpad;
  v8f acc[8];
#pragma unroll
  for (int j = 0; j < 8; ++j)
#pragma unroll
    for (int i = 0; i < 8; ++i) acc[j][i] = 0.0f;
  int nks = Kpad >> 5;
  v16bf a_cur = mkfrag(ld8(arow + half*8), ld8(arow + half*8 + 16));
  for (int ks = 0; ks < nks; ++ks){
    v16bf bfr[8];
#pragma unroll
    for (int j = 0; j < 8; ++j){
      const bf16_t* bp = Bt + (long)(j*16 + r) * Kpad + ks*32 + half*8;
      bfr[j] = mkfrag(ld8(bp), ld8(bp + 16));
    }
    v16bf a_nxt = a_cur;
    if (ks + 1 < nks){
      int ka = (ks+1)*32 + half*8;
      a_nxt = mkfrag(ld8(arow + ka), ld8(arow + ka + 16));
    }
#pragma unroll
    for (int j = 0; j < 8; ++j) acc[j] = wmma_bf16(a_cur, bfr[j], acc[j]);
    a_cur = a_nxt;
  }
#pragma unroll
  for (int j = 0; j < 8; ++j)
#pragma unroll
    for (int i = 0; i < 8; ++i){
      int m = half*8 + i, n = j*16 + r;
      smf[m*128 + n] = acc[j][i] + bias[n];
    }
  __syncthreads();
  if (l < 16){
    float s = 0.f, s2 = 0.f;
    for (int c = 0; c < 128; ++c){ float v = smf[l*128 + c]; s += v; s2 += v*v; }
    float mean = s * (1.0f/128.0f);
    float var  = s2 * (1.0f/128.0f) - mean*mean;
    float rstd = rsqrtf(var + 1e-5f);
    bf16_t* orow = out + (t*16 + l) * 128;
    for (int c = 0; c < 128; ++c){
      float x = (smf[l*128 + c] - mean) * rstd * g[c] + bb[c];
      orow[c] = (bf16_t)geluf(x);
    }
  }
}

__device__ __forceinline__ v8bf edge_chunk(const bf16_t* hs, const bf16_t* hd,
                                           const bf16_t* tl, int k){
  if (k < 128) return ld8(hs + k);
  if (k < 256) return ld8(hd + (k - 128));
  if (k < 264) return ld8(tl + (k - 256));
  v8bf z;
#pragma unroll
  for (int i = 0; i < 8; ++i) z[i] = (bf16_t)0.0f;
  return z;
}

// ---------------- edge MLP: 4 waves/block, LDS-staged weights ----------------
#define EW 4
__global__ __launch_bounds__(128)
void k_edge(const bf16_t* __restrict__ h, const bf16_t* __restrict__ tail,
            const int* __restrict__ src, const int* __restrict__ dst,
            const bf16_t* __restrict__ e1t, const float* __restrict__ e1b,
            const bf16_t* __restrict__ e2t, const float* __restrict__ e2b,
            float* __restrict__ hneigh){
  __shared__ bf16_t bsl[2][128*32];   // e1 weight slice double buffer (8KB each)
  __shared__ bf16_t be2[128*128];     // full e2 weights (32KB)
  __shared__ bf16_t sm[EW][16*128];   // per-wave m1 staging
  __shared__ int    sdst[EW][16];
  const int tid = threadIdx.x;
  const int wv = tid >> 5, l = tid & 31, r = l & 15, half = l >> 4;
  long t = (long)blockIdx.x * EW + wv;
  long e = t*16 + r;
  int se = src[e], de = dst[e];
  if (l < 16) sdst[wv][l] = de;
  const bf16_t* hs = h + (long)se * 128;
  const bf16_t* hd = h + (long)de * 128;
  const bf16_t* tl = tail + e * 8;

  // stage slice 0 of e1t, then all of e2t (column n = tid owns 64B / 256B)
  {
    const bf16_t* g1 = e1t + (long)tid * 288;
    bf16_t* l1 = &bsl[0][tid * 32];
    const bf16_t* g2 = e2t + (long)tid * 128;
    bf16_t* l2 = &be2[tid * 128];
#if USE_ASYNC
#pragma unroll
    for (int i = 0; i < 4; ++i)  async_cp16(g1 + i*8, l1 + i*8);
#pragma unroll
    for (int i = 0; i < 16; ++i) async_cp16(g2 + i*8, l2 + i*8);
    __builtin_amdgcn_s_wait_asynccnt(16);   // slice-0 (first 4, in-order) complete
#else
#pragma unroll
    for (int i = 0; i < 4; ++i)  *(v8bf*)(l1 + i*8) = ld8(g1 + i*8);
#pragma unroll
    for (int i = 0; i < 16; ++i) *(v8bf*)(l2 + i*8) = ld8(g2 + i*8);
#endif
  }
  __syncthreads();

  v8f acc[8];
#pragma unroll
  for (int j = 0; j < 8; ++j)
#pragma unroll
    for (int i = 0; i < 8; ++i) acc[j][i] = 0.0f;

  v16bf a_cur = mkfrag(edge_chunk(hs, hd, tl, half*8),
                       edge_chunk(hs, hd, tl, half*8 + 16));
  // GEMM1: K = 288 (9 slices of 32)
#pragma unroll
  for (int ks = 0; ks < 9; ++ks){
    if (ks < 8){ // prefetch next weight slice
      const bf16_t* g1 = e1t + (long)tid * 288 + (ks+1)*32;
      bf16_t* l1 = &bsl[(ks+1) & 1][tid * 32];
#if USE_ASYNC
#pragma unroll
      for (int i = 0; i < 4; ++i) async_cp16(g1 + i*8, l1 + i*8);
#else
#pragma unroll
      for (int i = 0; i < 4; ++i) *(v8bf*)(l1 + i*8) = ld8(g1 + i*8);
#endif
    }
    v16bf a_nxt = a_cur;
    if (ks < 8){
      int kn = (ks+1)*32 + half*8;
      a_nxt = mkfrag(edge_chunk(hs, hd, tl, kn), edge_chunk(hs, hd, tl, kn + 16));
    }
    const bf16_t* bb = &bsl[ks & 1][0];
    v16bf bfr[8];
#pragma unroll
    for (int j = 0; j < 8; ++j){
      const bf16_t* bp = bb + (j*16 + r)*32 + half*8;
      bfr[j] = mkfrag(ld8(bp), ld8(bp + 16));
    }
#pragma unroll
    for (int j = 0; j < 8; ++j) acc[j] = wmma_bf16(a_cur, bfr[j], acc[j]);
    a_cur = a_nxt;
#if USE_ASYNC
    __builtin_amdgcn_s_wait_asynccnt(0);
#endif
    __syncthreads();
  }

  // silu -> per-wave LDS staging (bf16 A for GEMM2)
#pragma unroll
  for (int j = 0; j < 8; ++j)
#pragma unroll
    for (int i = 0; i < 8; ++i){
      int m = half*8 + i, n = j*16 + r;
      sm[wv][m*128 + n] = (bf16_t)siluf(acc[j][i] + e1b[n]);
    }
  __syncthreads();

  // GEMM2: K = 128, A from LDS, B from be2
  v8f acc2[8];
#pragma unroll
  for (int j = 0; j < 8; ++j)
#pragma unroll
    for (int i = 0; i < 8; ++i) acc2[j][i] = 0.0f;
  const bf16_t* arow = &sm[wv][r * 128];
#pragma unroll
  for (int ks = 0; ks < 4; ++ks){
    int ka = ks*32 + half*8;
    v16bf a = mkfrag(ld8(arow + ka), ld8(arow + ka + 16));
    v16bf bfr[8];
#pragma unroll
    for (int j = 0; j < 8; ++j){
      const bf16_t* bp = &be2[(j*16 + r)*128 + ks*32 + half*8];
      bfr[j] = mkfrag(ld8(bp), ld8(bp + 16));
    }
#pragma unroll
    for (int j = 0; j < 8; ++j) acc2[j] = wmma_bf16(a, bfr[j], acc2[j]);
  }
#pragma unroll
  for (int j = 0; j < 8; ++j)
#pragma unroll
    for (int i = 0; i < 8; ++i){
      int m = half*8 + i, n = j*16 + r;
      float v = siluf(acc2[j][i] + e2b[n]);
      unsafeAtomicAdd(&hneigh[(long)sdst[wv][m] * 128 + n], v);
    }
}

// ---------------- node MLP ----------------
__global__ __launch_bounds__(32)
void k_node(const bf16_t* __restrict__ h, const bf16_t* __restrict__ hn,
            const bf16_t* __restrict__ n1t, const float* __restrict__ n1b,
            const bf16_t* __restrict__ n2t, const float* __restrict__ n2b,
            bf16_t* __restrict__ out, int do_ln,
            const float* __restrict__ g, const float* __restrict__ bb){
  __shared__ bf16_t sm[16*128];
  __shared__ float  smf[16*128];
  int l = threadIdx.x, r = l & 15, half = l >> 4;
  long t = blockIdx.x;
  const bf16_t* hrow  = h  + (t*16 + r) * 128;
  const bf16_t* hnrow = hn + (t*16 + r) * 128;

  v8f acc[8];
#pragma unroll
  for (int j = 0; j < 8; ++j)
#pragma unroll
    for (int i = 0; i < 8; ++i) acc[j][i] = 0.0f;

  v16bf a_cur = mkfrag(ld8(hrow + half*8), ld8(hrow + half*8 + 16));
#pragma unroll
  for (int ks = 0; ks < 8; ++ks){
    v16bf bfr[8];
#pragma unroll
    for (int j = 0; j < 8; ++j){
      const bf16_t* bp = n1t + (long)(j*16 + r) * 256 + ks*32 + half*8;
      bfr[j] = mkfrag(ld8(bp), ld8(bp + 16));
    }
    v16bf a_nxt = a_cur;
    if (ks < 7){
      const bf16_t* base = (ks+1 < 4) ? hrow : (hnrow - 128);
      int ka = (ks+1)*32 + half*8;
      a_nxt = mkfrag(ld8(base + ka), ld8(base + ka + 16));
    }
#pragma unroll
    for (int j = 0; j < 8; ++j) acc[j] = wmma_bf16(a_cur, bfr[j], acc[j]);
    a_cur = a_nxt;
  }
#pragma unroll
  for (int j = 0; j < 8; ++j)
#pragma unroll
    for (int i = 0; i < 8; ++i){
      int m = half*8 + i, n = j*16 + r;
      sm[m*128 + n] = (bf16_t)siluf(acc[j][i] + n1b[n]);
    }
  __syncthreads();

  v8f acc2[8];
#pragma unroll
  for (int j = 0; j < 8; ++j)
#pragma unroll
    for (int i = 0; i < 8; ++i) acc2[j][i] = 0.0f;
  const bf16_t* arow = sm + r * 128;
#pragma unroll
  for (int ks = 0; ks < 4; ++ks){
    int ka = ks*32 + half*8;
    v16bf a = mkfrag(ld8(arow + ka), ld8(arow + ka + 16));
    v16bf bfr[8];
#pragma unroll
    for (int j = 0; j < 8; ++j){
      const bf16_t* bp = n2t + (long)(j*16 + r) * 128 + ks*32 + half*8;
      bfr[j] = mkfrag(ld8(bp), ld8(bp + 16));
    }
#pragma unroll
    for (int j = 0; j < 8; ++j) acc2[j] = wmma_bf16(a, bfr[j], acc2[j]);
  }
  if (do_ln){
#pragma unroll
    for (int j = 0; j < 8; ++j)
#pragma unroll
      for (int i = 0; i < 8; ++i){
        int m = half*8 + i, n = j*16 + r;
        smf[m*128 + n] = acc2[j][i] + n2b[n];
      }
    __syncthreads();
    if (l < 16){
      float s = 0.f, s2 = 0.f;
      for (int c = 0; c < 128; ++c){ float v = smf[l*128 + c]; s += v; s2 += v*v; }
      float mean = s * (1.0f/128.0f);
      float var  = s2 * (1.0f/128.0f) - mean*mean;
      float rstd = rsqrtf(var + 1e-5f);
      bf16_t* orow = out + (t*16 + l) * 128;
      for (int c = 0; c < 128; ++c)
        orow[c] = (bf16_t)((smf[l*128 + c] - mean) * rstd * g[c] + bb[c]);
    }
  } else {
#pragma unroll
    for (int j = 0; j < 8; ++j)
#pragma unroll
      for (int i = 0; i < 8; ++i){
        int m = half*8 + i, n = j*16 + r;
        out[(t*16 + m) * 128 + n] = (bf16_t)(acc2[j][i] + n2b[n]);
      }
  }
}

// ---------------- heads + reparameterize ----------------
__global__ __launch_bounds__(32)
void k_heads(const bf16_t* __restrict__ h,
             const bf16_t* __restrict__ mut, const float* __restrict__ mub,
             const bf16_t* __restrict__ lvt, const float* __restrict__ lvb,
             const float* __restrict__ eps, float* __restrict__ out){
  int l = threadIdx.x, r = l & 15, half = l >> 4;
  long t = blockIdx.x;
  const bf16_t* arow = h + (t*16 + r) * 128;
  v8f am[4], al[4];
#pragma unroll
  for (int j = 0; j < 4; ++j)
#pragma unroll
    for (int i = 0; i < 8; ++i){ am[j][i] = 0.0f; al[j][i] = 0.0f; }
  v16bf a_cur = mkfrag(ld8(arow + half*8), ld8(arow + half*8 + 16));
#pragma unroll
  for (int ks = 0; ks < 4; ++ks){
    v16bf bfr[8];
#pragma unroll
    for (int j = 0; j < 4; ++j){
      const bf16_t* bm = mut + (long)(j*16 + r) * 128 + ks*32 + half*8;
      bfr[j] = mkfrag(ld8(bm), ld8(bm + 16));
      const bf16_t* bl = lvt + (long)(j*16 + r) * 128 + ks*32 + half*8;
      bfr[j+4] = mkfrag(ld8(bl), ld8(bl + 16));
    }
    v16bf a_nxt = a_cur;
    if (ks < 3){
      int ka = (ks+1)*32 + half*8;
      a_nxt = mkfrag(ld8(arow + ka), ld8(arow + ka + 16));
    }
#pragma unroll
    for (int j = 0; j < 4; ++j){
      am[j] = wmma_bf16(a_cur, bfr[j],   am[j]);
      al[j] = wmma_bf16(a_cur, bfr[j+4], al[j]);
    }
    a_cur = a_nxt;
  }
  const long NL = (long)N_NODES * 64;
#pragma unroll
  for (int j = 0; j < 4; ++j)
#pragma unroll
    for (int i = 0; i < 8; ++i){
      int m = half*8 + i, c = j*16 + r;
      long row = t*16 + m;
      float mu = am[j][i] + mub[c];
      float lv = al[j][i] + lvb[c];
      float z  = eps[row*64 + c] * __expf(lv) + mu;
      out[row*64 + c]        = z;
      out[NL + row*64 + c]   = mu;
      out[2*NL + row*64 + c] = lv;
    }
}

// ---------------- host orchestration ----------------

extern "C" void kernel_launch(void* const* d_in, const int* in_sizes, int n_in,
                              void* d_out, int out_size, void* d_ws, size_t ws_size,
                              hipStream_t stream){
  (void)in_sizes; (void)n_in; (void)out_size; (void)ws_size;
  const float* emb1_b = (const float*)d_in[0];
  const float* emb1_w = (const float*)d_in[1];
  const float* emb2_b = (const float*)d_in[2];
  const float* emb2_w = (const float*)d_in[3];
  const float* gcln_b = (const float*)d_in[4];
  const float* gcln_g = (const float*)d_in[5];
  const float* e1b[2] = {(const float*)d_in[6],  (const float*)d_in[14]};
  const float* e1w[2] = {(const float*)d_in[7],  (const float*)d_in[15]};
  const float* e2b[2] = {(const float*)d_in[8],  (const float*)d_in[16]};
  const float* e2w[2] = {(const float*)d_in[9],  (const float*)d_in[17]};
  const float* n1b[2] = {(const float*)d_in[10], (const float*)d_in[18]};
  const float* n1w[2] = {(const float*)d_in[11], (const float*)d_in[19]};
  const float* n2b[2] = {(const float*)d_in[12], (const float*)d_in[20]};
  const float* n2w[2] = {(const float*)d_in[13], (const float*)d_in[21]};
  const float* ln1_b = (const float*)d_in[22];
  const float* ln1_g = (const float*)d_in[23];
  const float* ln2_b = (const float*)d_in[24];
  const float* ln2_g = (const float*)d_in[25];
  const float* lv_b  = (const float*)d_in[26];
  const float* lv_w  = (const float*)d_in[27];
  const float* mu_b  = (const float*)d_in[28];
  const float* mu_w  = (const float*)d_in[29];
  const float* node_attr = (const float*)d_in[30];
  const float* edge_attr = (const float*)d_in[31];
  const float* xyz  = (const float*)d_in[32];
  const float* eps  = (const float*)d_in[33];
  const int* src = (const int*)d_in[34];
  const int* dst = (const int*)d_in[35];

  char* wsp = (char*)d_ws;
  auto carve = [&](size_t bytes)->char*{
    char* p = wsp; wsp += (bytes + 255) & ~(size_t)255; return p;
  };
  bf16_t* wt_emb1  = (bf16_t*)carve((size_t)96*128*2);
  bf16_t* wt_emb2  = (bf16_t*)carve((size_t)128*128*2);
  bf16_t* wt_e1[2] = {(bf16_t*)carve((size_t)288*128*2), (bf16_t*)carve((size_t)288*128*2)};
  bf16_t* wt_e2[2] = {(bf16_t*)carve((size_t)128*128*2), (bf16_t*)carve((size_t)128*128*2)};
  bf16_t* wt_n1[2] = {(bf16_t*)carve((size_t)256*128*2), (bf16_t*)carve((size_t)256*128*2)};
  bf16_t* wt_n2[2] = {(bf16_t*)carve((size_t)128*128*2), (bf16_t*)carve((size_t)128*128*2)};
  bf16_t* wt_mu = (bf16_t*)carve((size_t)128*64*2);
  bf16_t* wt_lv = (bf16_t*)carve((size_t)128*64*2);
  bf16_t* xpad  = (bf16_t*)carve((size_t)N_NODES*96*2);
  bf16_t* hA    = (bf16_t*)carve((size_t)N_NODES*128*2);
  bf16_t* hB    = (bf16_t*)carve((size_t)N_NODES*128*2);
  float*  hne   = (float*) carve((size_t)N_NODES*128*4);
  bf16_t* hneb  = (bf16_t*)carve((size_t)N_NODES*128*2);
  bf16_t* tail  = (bf16_t*)carve((size_t)N_EDGES*8*2);

  const int TB = 256;
#define GRID(n) ((int)(((n) + TB - 1) / TB))
  k_wcvt<<<GRID(96*128),  TB, 0, stream>>>(emb1_w, wt_emb1, 74, 96, 128);
  k_wcvt<<<GRID(128*128), TB, 0, stream>>>(emb2_w, wt_emb2, 128, 128, 128);
  for (int i = 0; i < 2; ++i){
    k_wcvt<<<GRID(288*128), TB, 0, stream>>>(e1w[i], wt_e1[i], 264, 288, 128);
    k_wcvt<<<GRID(128*128), TB, 0, stream>>>(e2w[i], wt_e2[i], 128, 128, 128);
    k_wcvt<<<GRID(256*128), TB, 0, stream>>>(n1w[i], wt_n1[i], 256, 256, 128);
    k_wcvt<<<GRID(128*128), TB, 0, stream>>>(n2w[i], wt_n2[i], 128, 128, 128);
  }
  k_wcvt<<<GRID(128*64), TB, 0, stream>>>(mu_w, wt_mu, 128, 128, 64);
  k_wcvt<<<GRID(128*64), TB, 0, stream>>>(lv_w, wt_lv, 128, 128, 64);

  k_pad_nodes<<<GRID((long)N_NODES*96), TB, 0, stream>>>(node_attr, xpad);
  k_tail<<<GRID(N_EDGES), TB, 0, stream>>>(xyz, edge_attr, src, dst, tail);

  const int node_tiles = N_NODES / 16;        // 3125
  const int edge_blocks = N_EDGES / 16 / EW;  // 12500

  k_gemm_ln_gelu<<<node_tiles, 32, 0, stream>>>(xpad, 96,  wt_emb1, emb1_b, ln1_g, ln1_b, hB);
  k_gemm_ln_gelu<<<node_tiles, 32, 0, stream>>>(hB,   128, wt_emb2, emb2_b, ln2_g, ln2_b, hA);

  bf16_t* hin = hA; bf16_t* hout = hB;
  for (int i = 0; i < 2; ++i){
    k_zero_f32<<<GRID((long)N_NODES*128), TB, 0, stream>>>(hne, (long)N_NODES*128);
    k_edge<<<edge_blocks, 128, 0, stream>>>(hin, tail, src, dst,
                                            wt_e1[i], e1b[i], wt_e2[i], e2b[i], hne);
    k_f32_to_bf16<<<GRID((long)N_NODES*128), TB, 0, stream>>>(hne, hneb, (long)N_NODES*128);
    k_node<<<node_tiles, 32, 0, stream>>>(hin, hneb, wt_n1[i], n1b[i], wt_n2[i], n2b[i],
                                          hout, (i == 0) ? 1 : 0, gcln_g, gcln_b);
    bf16_t* tmp = hin; hin = hout; hout = tmp;
  }

  k_heads<<<node_tiles, 32, 0, stream>>>(hin, wt_mu, mu_b, wt_lv, lv_b, eps, (float*)d_out);
#undef GRID
}
